// NunchakuQwenImageTransformerBlock_17669495455764
// MI455X (gfx1250) — matrix operand order repaired and checked by reading, not compiled, not run on hardware
//
#include <hip/hip_runtime.h>
#include <math.h>

#define DIM   3072
#define HEADS 24
#define HD    128
#define INNER 3072
#define FF    12288
#define S_IMG 1024
#define S_TXT 256
#define S_TOT 1280
#define EPS   1e-6f
#define ATT_SCALE 0.08838834764831845f   // 1/sqrt(128)

typedef __attribute__((ext_vector_type(16))) __bf16        v16bf;
typedef __attribute__((ext_vector_type(8)))  float         v8f;
typedef __attribute__((ext_vector_type(4)))  unsigned int  v4u;
typedef __attribute__((ext_vector_type(2)))  unsigned int  v2u;
typedef __attribute__((ext_vector_type(4)))  float         v4f;

union Frag16 { v16bf v; v4u q[2]; unsigned short u[16]; };

__device__ __forceinline__ unsigned short f2bf(float f) {
    unsigned int u = __float_as_uint(f);
    u += 0x7FFFu + ((u >> 16) & 1u);          // round-to-nearest-even
    return (unsigned short)(u >> 16);
}

// ---------------------------------------------------------------- silu(temb)
__global__ void silu_k(const float* __restrict__ temb, float* __restrict__ s) {
    int i = blockIdx.x * 256 + threadIdx.x;
    if (i < DIM) { float x = temb[i]; s[i] = x / (1.0f + __expf(-x)); }
}

// ------------------------------------------- mod GEMV: y = W@silu(temb)+b, reordered
__global__ void __launch_bounds__(256)
mod_gemv(const float* __restrict__ Wimg, const float* __restrict__ bimg,
         const float* __restrict__ Wtxt, const float* __restrict__ btxt,
         const float* __restrict__ s,
         float* __restrict__ img_mod, float* __restrict__ txt_mod)
{
    __shared__ float red[8];
    int r = blockIdx.x;                 // 0..36863
    int stream = r / (6 * DIM);
    int rr = r - stream * (6 * DIM);
    const float* W = stream ? Wtxt : Wimg;
    const float* B = stream ? btxt : bimg;
    float* out     = stream ? txt_mod : img_mod;
    const float* wrow = W + (size_t)rr * DIM;
    float acc = 0.f;
    for (int i = threadIdx.x; i < DIM; i += 256) acc += wrow[i] * s[i];
    #pragma unroll
    for (int m = 16; m >= 1; m >>= 1) acc += __shfl_xor(acc, m, 32);
    if ((threadIdx.x & 31) == 0) red[threadIdx.x >> 5] = acc;
    __syncthreads();
    if (threadIdx.x == 0) {
        float tot = 0.f;
        #pragma unroll
        for (int i = 0; i < 8; ++i) tot += red[i];
        int d = rr / 6, j = rr % 6;     // j: 0=shift1 1=scale1 2=gate1 3=shift2 4=scale2 5=gate2
        out[j * DIM + d] = tot + B[rr];
    }
}

// --------------------------------------------- LayerNorm + modulate -> bf16
__global__ void __launch_bounds__(256)
ln_mod(const float* __restrict__ X, const float* __restrict__ shift,
       const float* __restrict__ scale, unsigned short* __restrict__ Y)
{
    __shared__ float s1[8], s2[8];
    int t = blockIdx.x;
    const float* x = X + (size_t)t * DIM;
    float a = 0.f, b = 0.f;
    for (int i = threadIdx.x; i < DIM; i += 256) { float v = x[i]; a += v; b += v * v; }
    #pragma unroll
    for (int m = 16; m >= 1; m >>= 1) { a += __shfl_xor(a, m, 32); b += __shfl_xor(b, m, 32); }
    if ((threadIdx.x & 31) == 0) { s1[threadIdx.x >> 5] = a; s2[threadIdx.x >> 5] = b; }
    __syncthreads();
    float sa = 0.f, sb = 0.f;
    #pragma unroll
    for (int i = 0; i < 8; ++i) { sa += s1[i]; sb += s2[i]; }
    float mu = sa / DIM;
    float var = sb / DIM - mu * mu;
    float rs = rsqrtf(var + EPS);
    for (int i = threadIdx.x; i < DIM; i += 256) {
        float v = (x[i] - mu) * rs;
        Y[(size_t)t * DIM + i] = f2bf(v * (scale[i] + 1.0f) + shift[i]);
    }
}

// ---------------------------------------------------------- WMMA bf16 GEMM
// C[M,N] = A_bf16[M,K] * W_f32[N,K]^T + bias ; optional GELU ; f32 or bf16 out
// Double-buffered LDS; A tile staged with gfx1250 async global->LDS copies.
#define BM 128
#define BN 64
#define BK 32
#define LSTR 40   // ushort stride (pad vs bank conflicts)

template<int ACT, int OUTBF>
__global__ void __launch_bounds__(256)
gemm_wmma(const unsigned short* __restrict__ A, const float* __restrict__ W,
          const float* __restrict__ bias, void* __restrict__ Cout,
          int M, int N, int K)
{
    __shared__ __align__(16) unsigned short As[2][BM * LSTR];
    __shared__ __align__(16) unsigned short Bs[2][BN * LSTR];
    const int tid  = threadIdx.x;
    const int lane = tid & 31;
    const int wave = tid >> 5;
    const int wm = wave >> 1, wn = wave & 1;
    const int h = lane >> 4, l15 = lane & 15;
    const int m0 = blockIdx.y * BM;
    const int n0 = blockIdx.x * BN;

    auto stage = [&](int buf, int kb) {
        // A tile (bf16, 128x32): async global->LDS, ASYNCcnt-tracked
        #pragma unroll
        for (int i = 0; i < 2; ++i) {
            int c = tid + 256 * i;
            int r = c >> 2, c16 = c & 3;
            unsigned lds = (unsigned)(size_t)(&As[buf][r * LSTR + c16 * 8]);
            unsigned long long ga =
                (unsigned long long)(size_t)(A + (size_t)(m0 + r) * K + kb + c16 * 8);
            asm volatile("global_load_async_to_lds_b128 %0, %1, off"
                         :: "v"(lds), "v"(ga) : "memory");
        }
        // W tile (f32 -> bf16, 64x32): VALU convert, normal ds stores
        #pragma unroll
        for (int i = 0; i < 2; ++i) {
            int c = tid + 256 * i;
            int r = c >> 3, c4 = c & 7;
            const v4f wv = *(const v4f*)(W + (size_t)(n0 + r) * K + kb + c4 * 4);
            v2u pk;
            pk[0] = (unsigned)f2bf(wv[0]) | ((unsigned)f2bf(wv[1]) << 16);
            pk[1] = (unsigned)f2bf(wv[2]) | ((unsigned)f2bf(wv[3]) << 16);
            *(v2u*)(&Bs[buf][r * LSTR + c4 * 4]) = pk;
        }
        if (kb + BK < K)   // look-ahead prefetch of the W stream (global_prefetch_b8)
            __builtin_prefetch(W + (size_t)(n0 + (tid >> 2)) * K + kb + BK, 0, 1);
    };

    v8f acc[2][2];
    #pragma unroll
    for (int a = 0; a < 2; ++a)
        #pragma unroll
        for (int b = 0; b < 2; ++b)
            #pragma unroll
            for (int j = 0; j < 8; ++j) acc[a][b][j] = 0.f;

    stage(0, 0);
    asm volatile("s_wait_asynccnt 0x0" ::: "memory");
    __syncthreads();

    int buf = 0;
    for (int kb = 0; kb < K; kb += BK) {
        if (kb + BK < K) stage(buf ^ 1, kb + BK);   // overlap next-tile fetch with WMMAs

        Frag16 a[2], b[2];
        #pragma unroll
        for (int mt = 0; mt < 2; ++mt) {
            const unsigned short* p = &As[buf][(32 * wm + 16 * mt + l15) * LSTR];
            a[mt].q[0] = *(const v4u*)(p + 8 * h);        // K 0..7  / 8..15
            a[mt].q[1] = *(const v4u*)(p + 16 + 8 * h);   // K 16..23 / 24..31
        }
        #pragma unroll
        for (int nt = 0; nt < 2; ++nt) {
            const unsigned short* p = &Bs[buf][(32 * wn + 16 * nt + l15) * LSTR + 8 * h];
            b[nt].q[0] = *(const v4u*)(p);
            b[nt].q[1] = *(const v4u*)(p + 16);
        }
        #pragma unroll
        for (int mt = 0; mt < 2; ++mt)
            #pragma unroll
            for (int nt = 0; nt < 2; ++nt)
                acc[mt][nt] = __builtin_amdgcn_wmma_f32_16x16x32_bf16(
                    false, a[mt].v, false, b[nt].v, (short)0, acc[mt][nt], false, false);

        asm volatile("s_wait_asynccnt 0x0" ::: "memory");
        __syncthreads();
        buf ^= 1;
    }

    #pragma unroll
    for (int mt = 0; mt < 2; ++mt)
        #pragma unroll
        for (int nt = 0; nt < 2; ++nt)
            #pragma unroll
            for (int j = 0; j < 8; ++j) {
                int row = m0 + 32 * wm + 16 * mt + j + 8 * h;
                int col = n0 + 32 * wn + 16 * nt + l15;
                float v = acc[mt][nt][j] + bias[col];
                if (ACT == 1) {
                    float x = v;
                    v = 0.5f * x * (1.0f + tanhf(0.7978845608028654f * (x + 0.044715f * x * x * x)));
                }
                if (OUTBF) ((unsigned short*)Cout)[(size_t)row * N + col] = f2bf(v);
                else       ((float*)Cout)[(size_t)row * N + col] = v;
            }
}

// ------------------------------ per-(token,head): RMS norm + RoPE + relayout
__global__ void __launch_bounds__(128)
qkv_post(const float* __restrict__ qkv, int catoff,
         const float* __restrict__ gq, const float* __restrict__ gk,
         const float* __restrict__ rcos, const float* __restrict__ rsin,
         unsigned short* __restrict__ Qb, unsigned short* __restrict__ Kb,
         unsigned short* __restrict__ Vt)
{
    __shared__ float sq[4], sk[4];
    int t = blockIdx.x, head = blockIdx.y, d = threadIdx.x;
    const float* base = qkv + (size_t)t * (3 * INNER);
    float q = base[head * HD + d];
    float k = base[INNER + head * HD + d];
    float v = base[2 * INNER + head * HD + d];
    float pq = q * q, pk = k * k;
    #pragma unroll
    for (int m = 16; m >= 1; m >>= 1) { pq += __shfl_xor(pq, m, 32); pk += __shfl_xor(pk, m, 32); }
    if ((threadIdx.x & 31) == 0) { sq[threadIdx.x >> 5] = pq; sk[threadIdx.x >> 5] = pk; }
    __syncthreads();
    float msq = (sq[0] + sq[1] + sq[2] + sq[3]) * (1.0f / HD);
    float msk = (sk[0] + sk[1] + sk[2] + sk[3]) * (1.0f / HD);
    q = q * rsqrtf(msq + EPS) * gq[d];
    k = k * rsqrtf(msk + EPS) * gk[d];
    int pos = catoff + t;
    float c = rcos[pos * HD + d], s = rsin[pos * HD + d];
    float qn = __shfl_xor(q, 1, 32);
    float kn = __shfl_xor(k, 1, 32);
    float qr = (d & 1) ? qn : -qn;      // rot[2i]=-x[2i+1], rot[2i+1]=x[2i]
    float kr = (d & 1) ? kn : -kn;
    float qo = (q * c + qr * s) * ATT_SCALE;
    float ko = k * c + kr * s;
    int row = catoff + t;
    Qb[(size_t)head * S_TOT * HD + (size_t)row * HD + d] = f2bf(qo);
    Kb[(size_t)head * S_TOT * HD + (size_t)row * HD + d] = f2bf(ko);
    Vt[(size_t)head * HD * S_TOT + (size_t)d * S_TOT + row] = f2bf(v);
}

// ------------------------------------------- flash attention, 1 wave / 16 q-rows
__global__ void __launch_bounds__(32)
flash_attn(const unsigned short* __restrict__ Qb, const unsigned short* __restrict__ Kb,
           const unsigned short* __restrict__ Vt, unsigned short* __restrict__ AttnB)
{
    __shared__ __align__(16) unsigned short Plds[16 * LSTR];
    const int lane = threadIdx.x & 31;
    const int h = lane >> 4, l15 = lane & 15;
    const int q0 = blockIdx.x * 16;
    const int head = blockIdx.y;
    const unsigned short* Qh = Qb + (size_t)head * S_TOT * HD;
    const unsigned short* Kh = Kb + (size_t)head * S_TOT * HD;
    const unsigned short* Vh = Vt + (size_t)head * HD * S_TOT;

    Frag16 qf[4];
    #pragma unroll
    for (int s = 0; s < 4; ++s) {
        const unsigned short* p = Qh + (size_t)(q0 + l15) * HD + 32 * s + 16 * h;
        qf[s].q[0] = *(const v4u*)(p);
        qf[s].q[1] = *(const v4u*)(p + 8);
    }
    v8f o[8];
    float mrun[8], lrun[8];
    #pragma unroll
    for (int n = 0; n < 8; ++n)
        #pragma unroll
        for (int j = 0; j < 8; ++j) o[n][j] = 0.f;
    #pragma unroll
    for (int j = 0; j < 8; ++j) { mrun[j] = -1e30f; lrun[j] = 0.f; }

    for (int k0 = 0; k0 < S_TOT; k0 += 32) {
        v8f s0, s1;
        #pragma unroll
        for (int j = 0; j < 8; ++j) { s0[j] = 0.f; s1[j] = 0.f; }
        #pragma unroll
        for (int s = 0; s < 4; ++s) {
            Frag16 b0, b1;
            const unsigned short* p0 = Kh + (size_t)(k0 + l15) * HD + 32 * s + 16 * h;
            b0.q[0] = *(const v4u*)(p0);
            b0.q[1] = *(const v4u*)(p0 + 8);
            s0 = __builtin_amdgcn_wmma_f32_16x16x32_bf16(false, qf[s].v, false, b0.v, (short)0, s0, false, false);
            const unsigned short* p1 = Kh + (size_t)(k0 + 16 + l15) * HD + 32 * s + 16 * h;
            b1.q[0] = *(const v4u*)(p1);
            b1.q[1] = *(const v4u*)(p1 + 8);
            s1 = __builtin_amdgcn_wmma_f32_16x16x32_bf16(false, qf[s].v, false, b1.v, (short)0, s1, false, false);
        }
        // online softmax over 32 new keys (row = j + 8*h, cols across 16 lanes of half)
        float al[8];
        #pragma unroll
        for (int j = 0; j < 8; ++j) {
            float t = fmaxf(s0[j], s1[j]);
            #pragma unroll
            for (int m = 1; m < 16; m <<= 1) t = fmaxf(t, __shfl_xor(t, m, 32));
            float mn = fmaxf(mrun[j], t);
            float a  = __expf(mrun[j] - mn);
            float p0 = __expf(s0[j] - mn);
            float p1 = __expf(s1[j] - mn);
            float rs = p0 + p1;
            #pragma unroll
            for (int m = 1; m < 16; m <<= 1) rs += __shfl_xor(rs, m, 32);
            lrun[j] = lrun[j] * a + rs;
            mrun[j] = mn;
            al[j] = a;
            s0[j] = p0; s1[j] = p1;
        }
        #pragma unroll
        for (int n = 0; n < 8; ++n)
            #pragma unroll
            for (int j = 0; j < 8; ++j) o[n][j] = o[n][j] * al[j];
        // relayout P: D-layout -> A-layout via LDS
        __syncthreads();
        #pragma unroll
        for (int j = 0; j < 8; ++j) {
            Plds[(j + 8 * h) * LSTR + l15]      = f2bf(s0[j]);
            Plds[(j + 8 * h) * LSTR + 16 + l15] = f2bf(s1[j]);
        }
        __syncthreads();
        Frag16 pa;
        const unsigned short* pp = Plds + l15 * LSTR;
        pa.q[0] = *(const v4u*)(pp + 16 * h);
        pa.q[1] = *(const v4u*)(pp + 16 * h + 8);
        #pragma unroll
        for (int n = 0; n < 8; ++n) {
            Frag16 bv;
            const unsigned short* pv = Vh + (size_t)(16 * n + l15) * S_TOT + k0 + 16 * h;
            bv.q[0] = *(const v4u*)(pv);
            bv.q[1] = *(const v4u*)(pv + 8);
            o[n] = __builtin_amdgcn_wmma_f32_16x16x32_bf16(false, pa.v, false, bv.v, (short)0, o[n], false, false);
        }
    }
    #pragma unroll
    for (int n = 0; n < 8; ++n)
        #pragma unroll
        for (int j = 0; j < 8; ++j) {
            int row = q0 + j + 8 * h;
            int col = head * HD + 16 * n + l15;
            AttnB[(size_t)row * INNER + col] = f2bf(o[n][j] / lrun[j]);
        }
}

// --------------------------------------------------- out = X + gate (.) P
__global__ void resid_gate(float* __restrict__ Out, const float* __restrict__ X,
                           const float* __restrict__ gate, const float* __restrict__ P, int n)
{
    int i = blockIdx.x * 256 + threadIdx.x;
    if (i < n) { int d = i % DIM; Out[i] = X[i] + gate[d] * P[i]; }
}

// ===========================================================================
extern "C" void kernel_launch(void* const* d_in, const int* in_sizes, int n_in,
                              void* d_out, int out_size, void* d_ws, size_t ws_size,
                              hipStream_t stream)
{
    (void)in_sizes; (void)n_in; (void)out_size; (void)ws_size;
    const float* hidden    = (const float*)d_in[0];
    const float* enc       = (const float*)d_in[1];
    const float* temb      = (const float*)d_in[2];
    const float* rcos      = (const float*)d_in[3];
    const float* rsin      = (const float*)d_in[4];
    const float* img_mod_w = (const float*)d_in[5];
    const float* img_mod_b = (const float*)d_in[6];
    const float* txt_mod_w = (const float*)d_in[7];
    const float* txt_mod_b = (const float*)d_in[8];
    const float* qkv_w     = (const float*)d_in[9];
    const float* qkv_b     = (const float*)d_in[10];
    const float* aqkv_w    = (const float*)d_in[11];
    const float* aqkv_b    = (const float*)d_in[12];
    const float* norm_q_g  = (const float*)d_in[13];
    const float* norm_k_g  = (const float*)d_in[14];
    const float* norm_aq_g = (const float*)d_in[15];
    const float* norm_ak_g = (const float*)d_in[16];
    const float* to_out_w  = (const float*)d_in[17];
    const float* to_out_b  = (const float*)d_in[18];
    const float* to_aout_w = (const float*)d_in[19];
    const float* to_aout_b = (const float*)d_in[20];
    const float* i_mlp_w1  = (const float*)d_in[21];
    const float* i_mlp_b1  = (const float*)d_in[22];
    const float* i_mlp_w2  = (const float*)d_in[23];
    const float* i_mlp_b2  = (const float*)d_in[24];
    const float* t_mlp_w1  = (const float*)d_in[25];
    const float* t_mlp_b1  = (const float*)d_in[26];
    const float* t_mlp_w2  = (const float*)d_in[27];
    const float* t_mlp_b2  = (const float*)d_in[28];

    char* wsp = (char*)d_ws;
    size_t off = 0;
    auto carve = [&](size_t bytes) -> void* {
        void* r = wsp + off;
        off += (bytes + 255) & ~(size_t)255;
        return r;
    };
    float*          silu_v = (float*)carve((size_t)DIM * 4);
    float*          imod   = (float*)carve((size_t)6 * DIM * 4);
    float*          tmod   = (float*)carve((size_t)6 * DIM * 4);
    unsigned short* Xi     = (unsigned short*)carve((size_t)S_IMG * DIM * 2);
    unsigned short* Xt     = (unsigned short*)carve((size_t)S_TXT * DIM * 2);
    float*          QKVi   = (float*)carve((size_t)S_IMG * 3 * INNER * 4);
    float*          QKVt   = (float*)carve((size_t)S_TXT * 3 * INNER * 4);
    unsigned short* Qb     = (unsigned short*)carve((size_t)HEADS * S_TOT * HD * 2);
    unsigned short* Kb     = (unsigned short*)carve((size_t)HEADS * S_TOT * HD * 2);
    unsigned short* Vt     = (unsigned short*)carve((size_t)HEADS * HD * S_TOT * 2);
    unsigned short* Attn   = (unsigned short*)carve((size_t)S_TOT * INNER * 2);
    float*          Pi     = (float*)carve((size_t)S_IMG * DIM * 4);
    float*          Pt     = (float*)carve((size_t)S_TXT * DIM * 4);
    float*          Hi     = (float*)carve((size_t)S_IMG * DIM * 4);
    float*          Ht     = (float*)carve((size_t)S_TXT * DIM * 4);
    unsigned short* Mi1    = (unsigned short*)carve((size_t)S_IMG * FF * 2);
    unsigned short* Mt1    = (unsigned short*)carve((size_t)S_TXT * FF * 2);
    float*          Mi2    = (float*)carve((size_t)S_IMG * DIM * 4);
    float*          Mt2    = (float*)carve((size_t)S_TXT * DIM * 4);

    float* out_txt = (float*)d_out;
    float* out_img = out_txt + (size_t)S_TXT * DIM;

    // 1. modulation vectors
    silu_k<<<(DIM + 255) / 256, 256, 0, stream>>>(temb, silu_v);
    mod_gemv<<<2 * 6 * DIM, 256, 0, stream>>>(img_mod_w, img_mod_b, txt_mod_w, txt_mod_b,
                                              silu_v, imod, tmod);
    // 2. LN + modulate (phase 1)
    ln_mod<<<S_IMG, 256, 0, stream>>>(hidden, imod + 0 * DIM, imod + 1 * DIM, Xi);
    ln_mod<<<S_TXT, 256, 0, stream>>>(enc,    tmod + 0 * DIM, tmod + 1 * DIM, Xt);
    // 3. QKV projections
    gemm_wmma<0, 0><<<dim3(3 * INNER / BN, S_IMG / BM), 256, 0, stream>>>(
        Xi, qkv_w, qkv_b, QKVi, S_IMG, 3 * INNER, DIM);
    gemm_wmma<0, 0><<<dim3(3 * INNER / BN, S_TXT / BM), 256, 0, stream>>>(
        Xt, aqkv_w, aqkv_b, QKVt, S_TXT, 3 * INNER, DIM);
    // 4. RMS norm + RoPE + attention-layout
    qkv_post<<<dim3(S_IMG, HEADS), 128, 0, stream>>>(QKVi, S_TXT, norm_q_g, norm_k_g,
                                                     rcos, rsin, Qb, Kb, Vt);
    qkv_post<<<dim3(S_TXT, HEADS), 128, 0, stream>>>(QKVt, 0, norm_aq_g, norm_ak_g,
                                                     rcos, rsin, Qb, Kb, Vt);
    // 5. attention
    flash_attn<<<dim3(S_TOT / 16, HEADS), 32, 0, stream>>>(Qb, Kb, Vt, Attn);
    // 6. output projections
    gemm_wmma<0, 0><<<dim3(DIM / BN, S_IMG / BM), 256, 0, stream>>>(
        Attn + (size_t)S_TXT * INNER, to_out_w, to_out_b, Pi, S_IMG, DIM, INNER);
    gemm_wmma<0, 0><<<dim3(DIM / BN, S_TXT / BM), 256, 0, stream>>>(
        Attn, to_aout_w, to_aout_b, Pt, S_TXT, DIM, INNER);
    // 7. gated residual 1
    resid_gate<<<(S_IMG * DIM + 255) / 256, 256, 0, stream>>>(Hi, hidden, imod + 2 * DIM, Pi, S_IMG * DIM);
    resid_gate<<<(S_TXT * DIM + 255) / 256, 256, 0, stream>>>(Ht, enc,    tmod + 2 * DIM, Pt, S_TXT * DIM);
    // 8. LN + modulate (phase 2)
    ln_mod<<<S_IMG, 256, 0, stream>>>(Hi, imod + 3 * DIM, imod + 4 * DIM, Xi);
    ln_mod<<<S_TXT, 256, 0, stream>>>(Ht, tmod + 3 * DIM, tmod + 4 * DIM, Xt);
    // 9. MLP
    gemm_wmma<1, 1><<<dim3(FF / BN, S_IMG / BM), 256, 0, stream>>>(
        Xi, i_mlp_w1, i_mlp_b1, Mi1, S_IMG, FF, DIM);
    gemm_wmma<1, 1><<<dim3(FF / BN, S_TXT / BM), 256, 0, stream>>>(
        Xt, t_mlp_w1, t_mlp_b1, Mt1, S_TXT, FF, DIM);
    gemm_wmma<0, 0><<<dim3(DIM / BN, S_IMG / BM), 256, 0, stream>>>(
        Mi1, i_mlp_w2, i_mlp_b2, Mi2, S_IMG, DIM, FF);
    gemm_wmma<0, 0><<<dim3(DIM / BN, S_TXT / BM), 256, 0, stream>>>(
        Mt1, t_mlp_w2, t_mlp_b2, Mt2, S_TXT, DIM, FF);
    // 10. gated residual 2 -> outputs (txt first, then img)
    resid_gate<<<(S_IMG * DIM + 255) / 256, 256, 0, stream>>>(out_img, Hi, imod + 5 * DIM, Mi2, S_IMG * DIM);
    resid_gate<<<(S_TXT * DIM + 255) / 256, 256, 0, stream>>>(out_txt, Ht, tmod + 5 * DIM, Mt2, S_TXT * DIM);
}
